// PLCCLoss_40690520162753
// MI455X (gfx1250) — matrix operand order repaired and checked by reading, not compiled, not run on hardware
//
#include <hip/hip_runtime.h>
#include <math.h>

// ---------------------------------------------------------------------------
// PLCC contrastive loss, fused for MI455X (gfx1250, wave32, WMMA).
//   sim matmul  : v_wmma_f32_16x16x32_f16, K=64 in 2 k-tiles
//   pipelining  : B fragments + column coords register-double-buffered; the
//                 long scalar phase (dist/exp/top-k) hides global latency
//   distances   : scalar fp32 (K=3), coords L2-resident as float4 (xyz,|c|^2)
//   top-k sum   : streaming per-row top-KMAX LDS buffer + exact k-selection
// ---------------------------------------------------------------------------

typedef __attribute__((ext_vector_type(16))) _Float16 v16h;
typedef __attribute__((ext_vector_type(8)))  _Float16 v8h;
typedef __attribute__((ext_vector_type(2)))  _Float16 v2h;
typedef __attribute__((ext_vector_type(8)))  float    v8f;

#define KMAX      128          // >= k = 2*max_pos (realistic max ~50)
#define MAXB      5000         // MAX_BATCH_SIZE
#define TEMP_INV  10.0f        // 1 / TEMPERATURE
#define NEG_INF  (-3.0e38f)

static __device__ __forceinline__ void wave_wait_ds() {
    __asm__ volatile("s_wait_dscnt 0x0" ::: "memory");
}

// --------------------------- init accumulators -----------------------------
__global__ void plcc_init(float* acc, int* maxcnt, int nb) {
    int i = threadIdx.x;
    if (i < 2)  acc[i] = 0.0f;
    if (i < nb) maxcnt[i] = 0;
}

// ---- normalize rows -> fp16, and pack coords as (x,y,z,|c|^2) float4 ------
__global__ void plcc_prep(const float* __restrict__ feat,
                          const float* __restrict__ coords,
                          _Float16* __restrict__ fnorm,
                          float4*   __restrict__ cd, int N) {
    int wave = (int)((blockIdx.x * blockDim.x + threadIdx.x) >> 5);
    int lane = threadIdx.x & 31;
    if (wave >= N) return;
    const float* row = feat + (size_t)wave * 64;
    float2 v = *(const float2*)(row + lane * 2);
    float ss = v.x * v.x + v.y * v.y;
    #pragma unroll
    for (int m = 16; m >= 1; m >>= 1) ss += __shfl_xor(ss, m, 32);
    float inv = 1.0f / fmaxf(sqrtf(ss), 1e-8f);   // torch eps clamp
    v2h o; o[0] = (_Float16)(v.x * inv); o[1] = (_Float16)(v.y * inv);
    *(v2h*)(fnorm + (size_t)wave * 64 + lane * 2) = o;
    if (lane == 0) {
        float x = coords[3 * (size_t)wave + 0];
        float y = coords[3 * (size_t)wave + 1];
        float z = coords[3 * (size_t)wave + 2];
        cd[wave] = make_float4(x, y, z, x * x + y * y + z * z);
    }
}

// ---- per-row positive counts -> per-batch max (defines k) -----------------
__global__ void plcc_poscnt(const float4* __restrict__ cd,
                            int* __restrict__ maxcnt, int N) {
    int wave = (int)((blockIdx.x * blockDim.x + threadIdx.x) >> 5);
    int lane = threadIdx.x & 31;
    if (wave >= N) return;
    float4 r = cd[wave];
    int cnt = 0;
    for (int n = lane; n < N; n += 32) {
        float4 c = cd[n];
        float d2 = r.w + c.w - 2.0f * (r.x * c.x + r.y * c.y + r.z * c.z);
        // dist<1 <=> d2<1 ; dist>1e-6 <=> d2>1e-12
        cnt += (d2 < 1.0f && d2 > 1e-12f) ? 1 : 0;
    }
    #pragma unroll
    for (int m = 16; m >= 1; m >>= 1) cnt += __shfl_xor(cnt, m, 32);
    if (lane == 0) atomicMax(&maxcnt[wave / MAXB], cnt);
}

// --------------------------- fused main kernel -----------------------------
// One wave32 per 16-row tile; sweeps all N columns in 16-wide WMMA tiles.
__global__ __launch_bounds__(32)
void plcc_main(const _Float16* __restrict__ fnorm,
               const float4*   __restrict__ cd,
               const int*      __restrict__ maxcnt,
               float*          __restrict__ acc, int N) {
    __shared__ float  tile[16][17];         // staged 16x16 C fragment
    __shared__ float4 colc[16];             // column coords for current tile
    __shared__ float  topk[32 * KMAX];      // per-lane streaming top-K buffers

    const int lane  = threadIdx.x;          // blockDim.x == 32
    const int half  = lane >> 4;            // column half owned in scalar phase
    const int l16   = lane & 15;
    const int rbase = blockIdx.x * 16;
    const int myrow = rbase + l16;          // row owned in scalar phase
    const bool rowvalid = (myrow < N);

    // ---- A fragments for the wave's 16 rows (2 k-tiles of K=32) ----------
    // lane<16 : row=l16, K {0..7,16..23}(+32t); lane>=16: K {8..15,24..31}
    const _Float16* ap = fnorm + (size_t)min(myrow, N - 1) * 64;
    v16h afr[2];
    #pragma unroll
    for (int t = 0; t < 2; ++t) {
        const int ka = 32 * t + (half ? 8 : 0);
        v8h lo = *(const v8h*)(ap + ka);
        v8h hi = *(const v8h*)(ap + ka + 16);
        #pragma unroll
        for (int i = 0; i < 8; ++i) { afr[t][i] = lo[i]; afr[t][8 + i] = hi[i]; }
    }

    const float4 rc = cd[min(myrow, N - 1)];

    // init lane-private top-K buffer
    float* mybuf = &topk[(size_t)lane * KMAX];
    for (int q = 0; q < KMAX; ++q) mybuf[q] = NEG_INF;
    int   bcnt = 0, bmi = 0;
    float bmin = 3.0e38f;

    float pospart = 0.0f, contpart = 0.0f;

    // ---- pipeline prologue: B fragments + col coords for tile 0 ----------
    // B layout: lane<16: col=l16, K {0..7,16..23} (+32t); lane>=16: K {8..15,24..31} (+32t)
    const int kb0 = half ? 8 : 0;
    const _Float16* bp = fnorm + (size_t)min(l16, N - 1) * 64;
    v8h blo0 = *(const v8h*)(bp + kb0);
    v8h bhi0 = *(const v8h*)(bp + kb0 + 16);
    v8h blo1 = *(const v8h*)(bp + 32 + kb0);
    v8h bhi1 = *(const v8h*)(bp + 32 + kb0 + 16);
    float4 colreg = cd[min(l16, N - 1)];    // meaningful in lanes 0..15

    const int ntiles = (N + 15) >> 4;
    for (int tc = 0; tc < ntiles; ++tc) {
        const int cbase = tc * 16;

        // stage this tile's column coords (loaded last iteration)
        if (lane < 16) colc[lane] = colreg;

        // ---- WMMA: consume double-buffered B fragments -------------------
        v16h bfr0, bfr1;
        #pragma unroll
        for (int i = 0; i < 8; ++i) {
            bfr0[i] = blo0[i]; bfr0[8 + i] = bhi0[i];
            bfr1[i] = blo1[i]; bfr1[8 + i] = bhi1[i];
        }
        v8f c = {};
        c = __builtin_amdgcn_wmma_f32_16x16x32_f16(
                false, afr[0], false, bfr0, (short)0, c, false, false);
        c = __builtin_amdgcn_wmma_f32_16x16x32_f16(
                false, afr[1], false, bfr1, (short)0, c, false, false);

        // ---- issue next tile's loads; scalar phase hides their latency ---
        if (tc + 1 < ntiles) {
            const int nb = cbase + 16;
            const _Float16* bpn = fnorm + (size_t)min(nb + l16, N - 1) * 64;
            blo0 = *(const v8h*)(bpn + kb0);
            bhi0 = *(const v8h*)(bpn + kb0 + 16);
            blo1 = *(const v8h*)(bpn + 32 + kb0);
            bhi1 = *(const v8h*)(bpn + 32 + kb0 + 16);
            colreg = cd[min(nb + l16, N - 1)];
            if (nb + 16 < N)   // keep L1 warm two tiles ahead
                __builtin_prefetch(fnorm + (size_t)(nb + 16 + l16) * 64, 0, 3);
        }

        // ---- stage C tile to LDS ----------------------------------------
        #pragma unroll
        for (int r = 0; r < 8; ++r) tile[r + 8 * half][l16] = c[r];
        wave_wait_ds();   // wave-lockstep: all DS stores visible, no barrier

        // ---- scalar phase: lane owns (row=l16, cols 8*half..8*half+7) ----
        if (rowvalid) {
            #pragma unroll
            for (int j = 0; j < 8; ++j) {
                const int nc = cbase + 8 * half + j;
                if (nc < N) {
                    const float  s  = tile[l16][8 * half + j];
                    const float4 cc = colc[8 * half + j];
                    const float d2 = rc.w + cc.w -
                        2.0f * (rc.x * cc.x + rc.y * cc.y + rc.z * cc.z);
                    const bool pos = (d2 < 1.0f) && (d2 > 1e-12f);
                    if (pos) {
                        const float dist = sqrtf(fmaxf(d2, 0.0f));
                        pospart  += __expf(s * TEMP_INV);
                        contpart += fabsf(1.0f - s - dist);
                    } else {
                        // negative (self included): stream into top-K buffer
                        if (bcnt < KMAX) {
                            mybuf[bcnt] = s;
                            if (s < bmin) { bmin = s; bmi = bcnt; }
                            ++bcnt;
                        } else if (s > bmin) {
                            mybuf[bmi] = s;
                            float nm = 3.0e38f; int ni = 0;
                            for (int q = 0; q < KMAX; ++q) {
                                const float vv = mybuf[q];
                                if (vv < nm) { nm = vv; ni = q; }
                            }
                            bmin = nm; bmi = ni;
                        }
                    }
                }
            }
        }
    }

    // ---- combine the two column-halves of each row ------------------------
    const float posrow = pospart + __shfl_xor(pospart, 16, 32);
    wave_wait_ds();   // partner lane's buffer writes visible before selection

    float nce = 0.0f;
    if (lane < 16 && rowvalid) {
        const int mb = maxcnt[myrow / MAXB];
        int kk = 2 * mb;                       // NEG_SAMPLE_RATIO = 2.0
        kk = kk > N ? N : kk;
        kk = kk > 2 * KMAX ? 2 * KMAX : kk;
        float* b0 = &topk[(size_t)lane * KMAX];
        float* b1 = &topk[(size_t)(lane + 16) * KMAX];
        float ssum = 0.0f;
        for (int it = 0; it < kk; ++it) {      // exact top-k of the union
            float best = NEG_INF; int bi = 0, ws = 0;
            for (int q = 0; q < KMAX; ++q) {
                const float v0 = b0[q];
                if (v0 > best) { best = v0; bi = q; ws = 0; }
            }
            for (int q = 0; q < KMAX; ++q) {
                const float v1 = b1[q];
                if (v1 > best) { best = v1; bi = q; ws = 1; }
            }
            (ws ? b1 : b0)[bi] = NEG_INF;
            ssum += __expf(best * TEMP_INV);
        }
        nce = -__logf(posrow / (ssum + posrow + 1e-6f));
    }

    float contw = contpart;
    #pragma unroll
    for (int m = 16; m >= 1; m >>= 1) {
        nce   += __shfl_xor(nce,   m, 32);
        contw += __shfl_xor(contw, m, 32);
    }
    if (lane == 0) { atomicAdd(&acc[0], nce); atomicAdd(&acc[1], contw); }
}

// --------------------------- final combine ---------------------------------
__global__ void plcc_final(const float* __restrict__ acc,
                           float* __restrict__ out, int N) {
    const float M = (float)N;
    // loss = total_nce/M + GAMMA * total_cont/M ;  cont sum carries extra 1/M
    out[0] = acc[0] / M + 0.5f * (acc[1] / (M * M));
}

// ---------------------------------------------------------------------------
extern "C" void kernel_launch(void* const* d_in, const int* in_sizes, int n_in,
                              void* d_out, int out_size, void* d_ws, size_t ws_size,
                              hipStream_t stream) {
    const float* feat   = (const float*)d_in[0];
    const float* coords = (const float*)d_in[2];
    const int N  = in_sizes[1];               // labels count == rows
    const int NB = (N + MAXB - 1) / MAXB;

    // workspace layout (ws base is 256B aligned)
    char* w = (char*)d_ws;
    _Float16* fnorm = (_Float16*)w;
    size_t off = ((size_t)N * 64 * sizeof(_Float16) + 255) & ~(size_t)255;
    float4* cd = (float4*)(w + off);
    off += ((size_t)N * sizeof(float4) + 255) & ~(size_t)255;
    float* acc = (float*)(w + off);           // [0]=nce sum, [1]=cont sum
    int* maxcnt = (int*)(acc + 2);

    plcc_init<<<1, 256, 0, stream>>>(acc, maxcnt, NB);

    const int wavesBlocks = (N * 32 + 255) / 256;   // one wave per row
    plcc_prep  <<<wavesBlocks, 256, 0, stream>>>(feat, coords, fnorm, cd, N);
    plcc_poscnt<<<wavesBlocks, 256, 0, stream>>>(cd, maxcnt, N);

    plcc_main<<<(N + 15) / 16, 32, 0, stream>>>(fnorm, cd, maxcnt, acc, N);
    plcc_final<<<1, 1, 0, stream>>>(acc, (float*)d_out, N);
}
